// FullyParallelLlamaAttention_73693048865144
// MI455X (gfx1250) — matrix operand order repaired and checked by reading, not compiled, not run on hardware
//
#include <hip/hip_runtime.h>
#include <hip/hip_bf16.h>

// ---------------- CDNA5 WMMA types ----------------
typedef __attribute__((ext_vector_type(16))) __bf16 v16bf;
typedef __attribute__((ext_vector_type(8)))  __bf16 v8bf;
typedef __attribute__((ext_vector_type(8)))  float  v8f;

#define B_   2
#define S_   2048
#define D_   4096
#define H_   32
#define KV_  8
#define HD_  128
#define NREP_ (H_ / KV_)
#define SCALE_ 0.08838834764831845f   // 1/sqrt(128), folded into Q at RoPE
#define THETA_ 1000000.0f

// padded LDS row strides (halfwords) to spread 16-lane fragment reads over banks
#define KLDS_STRIDE 136   // 272 B per 32-key row of 256 B payload
#define VLDS_STRIDE 40    // 80 B per hd row of 64 B payload

__device__ __forceinline__ v16bf cat8(v8bf a, v8bf b) {
  return __builtin_shufflevector(a, b, 0,1,2,3,4,5,6,7,8,9,10,11,12,13,14,15);
}
__device__ __forceinline__ v16bf loadv16_16al(const __bf16* p) {  // 16B-aligned 32B load
  return cat8(*(const v8bf*)p, *(const v8bf*)(p + 8));
}
__device__ __forceinline__ v8f wmma_bf16(v16bf a, v16bf b, v8f c) {
  return __builtin_amdgcn_wmma_f32_16x16x32_bf16(false, a, false, b, (short)0, c, false, false);
}
__device__ __forceinline__ v8f zero8() {
  v8f z = {0.f,0.f,0.f,0.f,0.f,0.f,0.f,0.f};
  return z;
}

// generic (flat) pointer into a __shared__ object -> 32-bit LDS byte offset
typedef __attribute__((address_space(3))) void lds_void;
__device__ __forceinline__ unsigned int lds_addr_of(void* p) {
  return (unsigned int)(size_t)(lds_void*)p;
}
// CDNA5 async DMA: global -> LDS, 16B per lane, tracked by ASYNCcnt
__device__ __forceinline__ void async_g2l_b128(unsigned int lds_off, const void* gaddr) {
  unsigned long long g64 = (unsigned long long)(size_t)gaddr;
  asm volatile("global_load_async_to_lds_b128 %0, %1, off"
               :: "v"(lds_off), "v"(g64) : "memory");
}

// ---------------- elementwise f32 -> bf16 ----------------
__global__ __launch_bounds__(256) void k_f32_to_bf16(const float* __restrict__ in,
                                                     __bf16* __restrict__ out, int n) {
  int i = blockIdx.x * blockDim.x + threadIdx.x;
  if (i < n) out[i] = (__bf16)in[i];
}

// ---------------- f32 [R x C] -> bf16 [C x R] (LDS tile transpose) ----------------
__global__ __launch_bounds__(256) void k_transpose_f32_to_bf16(const float* __restrict__ in,
                                                               __bf16* __restrict__ out,
                                                               int R, int C) {
  __shared__ float tile[32][33];
  int c0 = blockIdx.x * 32, r0 = blockIdx.y * 32;
  int tx = threadIdx.x, ty = threadIdx.y;
  for (int i = ty; i < 32; i += 8)
    tile[i][tx] = in[(size_t)(r0 + i) * C + c0 + tx];
  __syncthreads();
  for (int i = ty; i < 32; i += 8)
    out[(size_t)(c0 + i) * R + r0 + tx] = (__bf16)tile[tx][i];
}

// ---------------- GEMM: C[M,N] = X[M,K](bf16) @ Wt[N,K](bf16)^T + bias ----------------
// 4 waves / WG; wave tile 32x64 (2x4 wmma tiles), WG tile 64x128.
__global__ __launch_bounds__(128) void k_gemm_bf16_wmma(
    const __bf16* __restrict__ X, const __bf16* __restrict__ Wt,
    const float* __restrict__ bias, float* __restrict__ Cout,
    int M, int N, int K) {
  const int lane = threadIdx.x & 31;
  const int wave = threadIdx.x >> 5;
  const int row  = lane & 15;
  const int hi   = lane >> 4;           // lane-half
  const int wm   = wave >> 1, wn = wave & 1;
  const int row0 = blockIdx.y * 64 + wm * 32;
  const int col0 = blockIdx.x * 128 + wn * 64;

  v8f acc[2][4];
#pragma unroll
  for (int i = 0; i < 2; ++i)
#pragma unroll
    for (int j = 0; j < 4; ++j) acc[i][j] = zero8();

  for (int k = 0; k < K; k += 32) {
    v16bf a[2], b[4];
#pragma unroll
    for (int i = 0; i < 2; ++i) {
      // A frag: lane-half 0 holds K k..k+15 ; half 1 holds k+16..k+31 (contiguous)
      const __bf16* p = X + (size_t)(row0 + i * 16 + row) * K + k + hi * 16;
      a[i] = cat8(*(const v8bf*)p, *(const v8bf*)(p + 8));
    }
#pragma unroll
    for (int j = 0; j < 4; ++j) {
      // B frag: lane-half 0 holds K k..k+15 ; half 1 holds k+16..k+31 (contiguous)
      const __bf16* p = Wt + (size_t)(col0 + j * 16 + row) * K + k + hi * 16;
      b[j] = *(const v16bf*)p;
    }
#pragma unroll
    for (int i = 0; i < 2; ++i)
#pragma unroll
      for (int j = 0; j < 4; ++j)
        acc[i][j] = wmma_bf16(a[i], b[j], acc[i][j]);
  }

#pragma unroll
  for (int i = 0; i < 2; ++i)
#pragma unroll
    for (int j = 0; j < 4; ++j) {
      int cg = col0 + j * 16 + row;
      float bv = bias ? bias[cg] : 0.0f;
#pragma unroll
      for (int r = 0; r < 8; ++r) {
        int rg = row0 + i * 16 + r + 8 * hi;     // C layout: row = r + 8*hi
        Cout[(size_t)rg * N + cg] = acc[i][j][r] + bv;
      }
    }
}

// ---------------- RoPE Q: f32 [B*S, H*HD] -> bf16 [B,H,S,HD], scaled ----------------
__global__ __launch_bounds__(256) void k_rope_q(const float* __restrict__ Qf,
                                                const int* __restrict__ pos_ids,
                                                __bf16* __restrict__ Qb) {
  int id = blockIdx.x * blockDim.x + threadIdx.x;
  if (id >= B_ * S_ * H_ * 64) return;
  int d0 = id & 63;
  int t = id >> 6;
  int h = t % H_; t /= H_;
  int s = t % S_;
  int b = t / S_;
  float pos = (float)pos_ids[b * S_ + s];
  float ang = pos * __powf(THETA_, -(float)d0 * (1.0f / 64.0f));
  float c = __cosf(ang), sn = __sinf(ang);
  const float* qr = Qf + (size_t)(b * S_ + s) * (H_ * HD_) + h * HD_;
  float x1 = qr[d0], x2 = qr[d0 + 64];
  __bf16* o = Qb + ((size_t)(b * H_ + h) * S_ + s) * HD_;
  o[d0]      = (__bf16)((x1 * c - x2 * sn) * SCALE_);
  o[d0 + 64] = (__bf16)((x1 * sn + x2 * c) * SCALE_);
}

// ---------------- RoPE K: f32 [B*S, KV*HD] -> bf16 [B,KV,S,HD] ----------------
__global__ __launch_bounds__(256) void k_rope_k(const float* __restrict__ Kf,
                                                const int* __restrict__ pos_ids,
                                                __bf16* __restrict__ Kb) {
  int id = blockIdx.x * blockDim.x + threadIdx.x;
  if (id >= B_ * S_ * KV_ * 64) return;
  int d0 = id & 63;
  int t = id >> 6;
  int h = t % KV_; t /= KV_;
  int s = t % S_;
  int b = t / S_;
  float pos = (float)pos_ids[b * S_ + s];
  float ang = pos * __powf(THETA_, -(float)d0 * (1.0f / 64.0f));
  float c = __cosf(ang), sn = __sinf(ang);
  const float* kr = Kf + (size_t)(b * S_ + s) * (KV_ * HD_) + h * HD_;
  float x1 = kr[d0], x2 = kr[d0 + 64];
  __bf16* o = Kb + ((size_t)(b * KV_ + h) * S_ + s) * HD_;
  o[d0]      = (__bf16)(x1 * c - x2 * sn);
  o[d0 + 64] = (__bf16)(x1 * sn + x2 * c);
}

// ---------------- V pack: f32 [B*S, KV*HD] -> bf16 [B,KV,HD,S] (transposed) ----------------
__global__ __launch_bounds__(256) void k_pack_vt(const float* __restrict__ Vf,
                                                 __bf16* __restrict__ Vt) {
  int id = blockIdx.x * blockDim.x + threadIdx.x;
  if (id >= B_ * KV_ * HD_ * S_) return;
  int s = id & (S_ - 1);
  int t = id >> 11;                   // /S_
  int hd = t % HD_; t /= HD_;
  int kv = t % KV_;
  int b  = t / KV_;
  Vt[id] = (__bf16)Vf[(size_t)(b * S_ + s) * (KV_ * HD_) + kv * HD_ + hd];
}

// ---------------- Flash attention (async double-buffered K/V in LDS) ----------------
// WG = 4 waves = 64 q rows of one (b,h). K/V tiles (32 keys) staged once per WG via
// global_load_async_to_lds_b128, double-buffered, shared by all 4 waves.
// Q bf16 [B,H,S,HD] (pre-scaled), K bf16 [B,KV,S,HD], Vt bf16 [B,KV,HD,S], O bf16 [B*S,H*HD]
__global__ __launch_bounds__(128) void k_flash_attn(
    const __bf16* __restrict__ Q, const __bf16* __restrict__ Km,
    const __bf16* __restrict__ Vt, __bf16* __restrict__ O) {
  __shared__ __align__(16) __bf16 sK[2][32 * KLDS_STRIDE];   // 2 x 8704 B
  __shared__ __align__(16) __bf16 sV[2][128 * VLDS_STRIDE];  // 2 x 10240 B
  __shared__ __align__(16) __bf16 sP[4][16 * 32];            // per-wave P staging

  const int tid  = threadIdx.x;
  const int lane = tid & 31;
  const int wave = tid >> 5;
  const int row  = lane & 15;
  const int hi   = lane >> 4;

  const int q64 = blockIdx.x & (S_ / 64 - 1);   // 64-row q tile of this WG
  const int bh  = blockIdx.x / (S_ / 64);
  const int h   = bh % H_;
  const int b   = bh / H_;
  const int kv  = h / NREP_;
  const int qt  = q64 * 4 + wave;               // this wave's 16-row tile

  const __bf16* Qp = Q  + ((size_t)(b * H_  + h)  * S_ + qt * 16) * HD_;
  const char*   Kg = (const char*)(Km + ((size_t)(b * KV_ + kv) * S_) * HD_);
  const char*   Vg = (const char*)(Vt + ((size_t)(b * KV_ + kv) * HD_) * S_);
  __bf16* pl = sP[wave];

  // Q A-fragments (16 rows x 128, four 16x32 chunks)
  v16bf qa[4];
#pragma unroll
  for (int c = 0; c < 4; ++c) {
    const __bf16* p = Qp + row * HD_ + c * 32 + hi * 16;
    qa[c] = cat8(*(const v8bf*)p, *(const v8bf*)(p + 8));
  }

  v8f acc[8];
#pragma unroll
  for (int j = 0; j < 8; ++j) acc[j] = zero8();
  float m_i[8], l_i[8];
#pragma unroll
  for (int r = 0; r < 8; ++r) { m_i[r] = -1e30f; l_i[r] = 0.0f; }

  const int nblocks = 2 * (q64 + 1);            // 32-key blocks up to WG causal limit

  // ---- async stage of one 32-key block into LDS buffer bp (8 loads per wave) ----
  auto stage = [&](int kb, int bp) {
#pragma unroll
    for (int i = 0; i < 4; ++i) {               // K: 32 rows x 256 B
      int chunk = tid + i * 128;                // 0..511
      int krow  = chunk >> 4;
      int koff  = (chunk & 15) * 16;
      async_g2l_b128(lds_addr_of((char*)sK[bp] + krow * (KLDS_STRIDE * 2) + koff),
                     Kg + (size_t)(kb * 32 + krow) * (HD_ * 2) + koff);
    }
#pragma unroll
    for (int i = 0; i < 4; ++i) {               // V: 128 rows x 64 B (row stride S*2)
      int chunk = tid + i * 128;
      int vrow  = chunk >> 2;
      int voff  = (chunk & 3) * 16;
      async_g2l_b128(lds_addr_of((char*)sV[bp] + vrow * (VLDS_STRIDE * 2) + voff),
                     Vg + (size_t)vrow * (S_ * 2) + kb * 64 + voff);
    }
  };

  stage(0, 0);
  for (int kb = 0; kb < nblocks; ++kb) {
    const int bp   = kb & 1;
    const int key0 = kb * 32;
    if (kb + 1 < nblocks) {
      stage(kb + 1, bp ^ 1);
      asm volatile("s_wait_asynccnt 8" ::: "memory");   // block kb's 8 loads done
    } else {
      asm volatile("s_wait_asynccnt 0" ::: "memory");
    }
    __syncthreads();                                    // all waves' staging visible

    if (key0 < qt * 16 + 16) {                          // wave-uniform: causal skip
      v8f s0 = zero8(), s1 = zero8();
#pragma unroll
      for (int c = 0; c < 4; ++c) {                     // QK^T, contract over HD
        const __bf16* kp0 = sK[bp] + (row)      * KLDS_STRIDE + c * 32 + hi * 16;
        const __bf16* kp1 = sK[bp] + (row + 16) * KLDS_STRIDE + c * 32 + hi * 16;
        s0 = wmma_bf16(qa[c], loadv16_16al(kp0), s0);
        s1 = wmma_bf16(qa[c], loadv16_16al(kp1), s1);
      }
      if (key0 + 31 > qt * 16) {                        // diagonal block: mask
#pragma unroll
        for (int r = 0; r < 8; ++r) {
          int qg = qt * 16 + r + 8 * hi;
          int c0 = key0 + row;
          s0[r] = (c0 > qg)      ? -1e30f : s0[r];
          s1[r] = (c0 + 16 > qg) ? -1e30f : s1[r];
        }
      }
      // online softmax; row = r + 8*hi, reduce across the 16-lane half
#pragma unroll
      for (int r = 0; r < 8; ++r) {
        float t = fmaxf(s0[r], s1[r]);
        t = fmaxf(t, __shfl_xor(t, 1, 32));
        t = fmaxf(t, __shfl_xor(t, 2, 32));
        t = fmaxf(t, __shfl_xor(t, 4, 32));
        t = fmaxf(t, __shfl_xor(t, 8, 32));
        float mn   = fmaxf(m_i[r], t);
        float corr = __expf(m_i[r] - mn);
        float e0 = __expf(s0[r] - mn);
        float e1 = __expf(s1[r] - mn);
        s0[r] = e0; s1[r] = e1;
        float rs = e0 + e1;
        rs += __shfl_xor(rs, 1, 32);
        rs += __shfl_xor(rs, 2, 32);
        rs += __shfl_xor(rs, 4, 32);
        rs += __shfl_xor(rs, 8, 32);
        l_i[r] = l_i[r] * corr + rs;
        m_i[r] = mn;
#pragma unroll
        for (int j = 0; j < 8; ++j) acc[j][r] *= corr;
      }
      // P (C layout f32) -> bf16 A-fragment via per-wave LDS slice (in-order per wave)
#pragma unroll
      for (int r = 0; r < 8; ++r) {
        int pr = r + 8 * hi;
        pl[pr * 32 + row]      = (__bf16)s0[r];
        pl[pr * 32 + 16 + row] = (__bf16)s1[r];
      }
      asm volatile("" ::: "memory");
      const __bf16* prd = pl + row * 32;
      v16bf pa = cat8(*(const v8bf*)(prd + hi * 16), *(const v8bf*)(prd + hi * 16 + 8));
      asm volatile("" ::: "memory");
      // PV: contract over 32 keys, 8 hd chunks
#pragma unroll
      for (int j = 0; j < 8; ++j) {
        const __bf16* vp = sV[bp] + (j * 16 + row) * VLDS_STRIDE + hi * 16;
        acc[j] = wmma_bf16(pa, loadv16_16al(vp), acc[j]);
      }
    }
    __syncthreads();   // buffer bp free to overwrite next iteration
  }

  // normalize and store O bf16 [B*S, H*HD]
#pragma unroll
  for (int r = 0; r < 8; ++r) {
    float inv = 1.0f / l_i[r];
    int sg = qt * 16 + r + 8 * hi;
    __bf16* op = O + (size_t)(b * S_ + sg) * (H_ * HD_) + h * HD_;
#pragma unroll
    for (int j = 0; j < 8; ++j)
      op[j * 16 + row] = (__bf16)(acc[j][r] * inv);
  }
}

// ---------------- host side ----------------
extern "C" void kernel_launch(void* const* d_in, const int* in_sizes, int n_in,
                              void* d_out, int out_size, void* d_ws, size_t ws_size,
                              hipStream_t stream) {
  const float* hidden = (const float*)d_in[0];
  const int*   pos    = (const int*)d_in[1];
  const float* wq = (const float*)d_in[2];
  const float* bq = (const float*)d_in[3];
  const float* wk = (const float*)d_in[4];
  const float* bk = (const float*)d_in[5];
  const float* wv = (const float*)d_in[6];
  const float* bv = (const float*)d_in[7];
  const float* wo = (const float*)d_in[8];
  float* out = (float*)d_out;

  char* ws = (char*)d_ws;
  size_t off = 0;
  auto alloc = [&](size_t bytes) {
    size_t o = off;
    off += (bytes + 255) & ~(size_t)255;
    return o;
  };
  const size_t M = (size_t)B_ * S_;                       // 4096 tokens
  __bf16* Xb  = (__bf16*)(ws + alloc(M * D_ * 2));        // hidden bf16
  __bf16* WqT = (__bf16*)(ws + alloc((size_t)D_ * D_ * 2));
  __bf16* WkT = (__bf16*)(ws + alloc((size_t)KV_ * HD_ * D_ * 2));
  __bf16* WvT = (__bf16*)(ws + alloc((size_t)KV_ * HD_ * D_ * 2));
  __bf16* WoT = (__bf16*)(ws + alloc((size_t)D_ * D_ * 2));
  float*  Qf  = (float*)(ws + alloc(M * (size_t)(H_ * HD_) * 4));
  float*  Kf  = (float*)(ws + alloc(M * (size_t)(KV_ * HD_) * 4));
  float*  Vf  = (float*)(ws + alloc(M * (size_t)(KV_ * HD_) * 4));
  __bf16* Qb  = (__bf16*)(ws + alloc((size_t)B_ * H_ * S_ * HD_ * 2));
  __bf16* Kbb = (__bf16*)(ws + alloc((size_t)B_ * KV_ * S_ * HD_ * 2));
  __bf16* Vtb = (__bf16*)(ws + alloc((size_t)B_ * KV_ * HD_ * S_ * 2));
  __bf16* Ob  = (__bf16*)Qf;   // reuse Qf region (free after RoPE) for attn output

  // 1. hidden f32 -> bf16
  int nX = (int)(M * D_);
  k_f32_to_bf16<<<(nX + 255) / 256, 256, 0, stream>>>(hidden, Xb, nX);

  // 2. weight transposes (f32 -> bf16, Wt[N,K])
  dim3 tb(32, 8);
  k_transpose_f32_to_bf16<<<dim3(4096 / 32, 4096 / 32), tb, 0, stream>>>(wq, WqT, 4096, 4096);
  k_transpose_f32_to_bf16<<<dim3(1024 / 32, 4096 / 32), tb, 0, stream>>>(wk, WkT, 4096, 1024);
  k_transpose_f32_to_bf16<<<dim3(1024 / 32, 4096 / 32), tb, 0, stream>>>(wv, WvT, 4096, 1024);
  k_transpose_f32_to_bf16<<<dim3(4096 / 32, 4096 / 32), tb, 0, stream>>>(wo, WoT, 4096, 4096);

  // 3. QKV projections (WMMA)
  k_gemm_bf16_wmma<<<dim3(4096 / 128, 4096 / 64), 128, 0, stream>>>(Xb, WqT, bq, Qf, 4096, 4096, 4096);
  k_gemm_bf16_wmma<<<dim3(1024 / 128, 4096 / 64), 128, 0, stream>>>(Xb, WkT, bk, Kf, 4096, 1024, 4096);
  k_gemm_bf16_wmma<<<dim3(1024 / 128, 4096 / 64), 128, 0, stream>>>(Xb, WvT, bv, Vf, 4096, 1024, 4096);

  // 4. RoPE + layout packing
  int nQ = B_ * S_ * H_ * 64;
  k_rope_q<<<(nQ + 255) / 256, 256, 0, stream>>>(Qf, pos, Qb);
  int nK = B_ * S_ * KV_ * 64;
  k_rope_k<<<(nK + 255) / 256, 256, 0, stream>>>(Kf, pos, Kbb);
  int nV = B_ * KV_ * HD_ * S_;
  k_pack_vt<<<(nV + 255) / 256, 256, 0, stream>>>(Vf, Vtb);

  // 5. causal flash attention (WMMA + async LDS staging)
  int nwg = B_ * H_ * (S_ / 64);           // 2048 WGs, 4 waves each
  k_flash_attn<<<nwg, 128, 0, stream>>>(Qb, Kbb, Vtb, Ob);

  // 6. output projection (WMMA)
  k_gemm_bf16_wmma<<<dim3(4096 / 128, 4096 / 64), 128, 0, stream>>>(Ob, WoT, nullptr, out, 4096, 4096, 4096);
}
